// SGC_74148315398479
// MI455X (gfx1250) — matrix operand order repaired and checked by reading, not compiled, not run on hardware
//
#include <hip/hip_runtime.h>
#include <stdint.h>

typedef __attribute__((ext_vector_type(16))) _Float16 v16h;
typedef __attribute__((ext_vector_type(8)))  float    v8f;

#define N_FEAT  256
#define N_CLASS 40

// ---------------- degree accumulation (segment_sum of w at col) ----------------
__global__ void sgc_deg_kernel(const long long* __restrict__ col,
                               const float* __restrict__ w,
                               float* __restrict__ deg, int E) {
    int e = blockIdx.x * blockDim.x + threadIdx.x;
    if (e < E) atomicAdd(&deg[col[e]], w[e]);
}

// deg -> dinv (in place, with +1 self loop) and selfnorm = dinv^2
__global__ void sgc_dinv_kernel(float* __restrict__ dinv,
                                float* __restrict__ selfnorm, int N) {
    int i = blockIdx.x * blockDim.x + threadIdx.x;
    if (i < N) {
        float d = dinv[i] + 1.0f;   // self-loop weight 1.0
        float r = rsqrtf(d);        // d >= 1 > 0 always
        dinv[i] = r;
        selfnorm[i] = r * r;
    }
}

// norm[e] = dinv[row]*w*dinv[col]
__global__ void sgc_norm_kernel(const long long* __restrict__ row,
                                const long long* __restrict__ col,
                                const float* __restrict__ w,
                                const float* __restrict__ dinv,
                                float* __restrict__ norm, int E) {
    int e = blockIdx.x * blockDim.x + threadIdx.x;
    if (e < E) norm[e] = dinv[row[e]] * w[e] * dinv[col[e]];
}

// ---------------- propagation: xout[col] += norm * xin[row] -------------------
// 64 threads per edge, float4 per thread (256 feats). Coalesced gather, f32
// atomic scatter (resolves in L2; x matrix is L2-resident on MI455X's 192MB L2).
__global__ void sgc_prop_edge_kernel(const long long* __restrict__ row,
                                     const long long* __restrict__ col,
                                     const float* __restrict__ norm,
                                     const float* __restrict__ xin,
                                     float* __restrict__ xout, int E) {
    long long t = (long long)blockIdx.x * blockDim.x + threadIdx.x;
    int e = (int)(t >> 6);
    int f = ((int)t & 63) << 2;
    if (e >= E) return;
    long long r = row[e];
    long long c = col[e];
    const float* src = xin + r * N_FEAT + f;
    __builtin_prefetch(src, 0, 3);          // global_prefetch_b8: start row fetch early
    float nrm = norm[e];
    float4 v = *(const float4*)src;
    float* dst = xout + c * N_FEAT + f;
    atomicAdd(dst + 0, nrm * v.x);
    atomicAdd(dst + 1, nrm * v.y);
    atomicAdd(dst + 2, nrm * v.z);
    atomicAdd(dst + 3, nrm * v.w);
}

// self loop: xout[i] += selfnorm[i]*xin[i]  (runs after edge kernel; no atomics needed)
__global__ void sgc_prop_self_kernel(const float* __restrict__ selfnorm,
                                     const float* __restrict__ xin,
                                     float* __restrict__ xout, int N) {
    long long t = (long long)blockIdx.x * blockDim.x + threadIdx.x;
    int i = (int)(t >> 6);
    int f = ((int)t & 63) << 2;
    if (i >= N) return;
    float s = selfnorm[i];
    const float4 v = *(const float4*)(xin + (long long)i * N_FEAT + f);
    float4* d = (float4*)(xout + (long long)i * N_FEAT + f);
    float4 o = *d;
    o.x += s * v.x; o.y += s * v.y; o.z += s * v.z; o.w += s * v.w;
    *d = o;
}

// ---------------- classifier GEMM via WMMA: out[r][c] = X[r]·W[c] + b[c] ------
// One wave32 per 16x16 output tile; K=256 -> 8x v_wmma_f32_16x16x32_f16.
// A/B operand packing follows the CDNA5 ISA 16-bit 16x32 / 32x16 VGPR layouts.
__global__ __launch_bounds__(32)
void sgc_gemm_wmma_kernel(const float* __restrict__ X,
                          const float* __restrict__ W,
                          const float* __restrict__ b,
                          float* __restrict__ out, int N) {
    int mtile = blockIdx.x / 3;
    int ntile = blockIdx.x % 3;   // 3 tiles cover 48 >= 40 classes
    int lane  = threadIdx.x;
    int half  = lane >> 4;        // 0: lanes 0-15, 1: lanes 16-31
    int m     = lane & 15;

    long long rowIdx = (long long)mtile * 16 + m;
    bool rowValid = rowIdx < N;
    int  cls      = ntile * 16 + m;
    bool clsValid = cls < N_CLASS;

    const float* xrow = X + rowIdx * N_FEAT;
    const float* wrow = W + (long long)cls * N_FEAT;

    v8f acc = {};
    #pragma unroll
    for (int k0 = 0; k0 < N_FEAT; k0 += 32) {
        v16h a, bb;
        // A (16x32 f16): lane half 0 holds K = k0+{0..7, 16..23}, half 1 K = k0+{8..15, 24..31}
        if (rowValid) {
            const float* p0 = xrow + k0 + half * 8;
            const float* p1 = xrow + k0 + 16 + half * 8;
            #pragma unroll
            for (int j = 0; j < 8; ++j) {
                a[j]     = (_Float16)p0[j];
                a[8 + j] = (_Float16)p1[j];
            }
        } else {
            #pragma unroll
            for (int j = 0; j < 16; ++j) a[j] = (_Float16)0.0f;
        }
        // B (32x16 f16): lane = column (class); half 0 holds K = k0+{0..7, 16..23}, half 1 K = k0+{8..15, 24..31}
        if (clsValid) {
            const float* q = wrow + k0 + half * 8;
            #pragma unroll
            for (int j = 0; j < 16; ++j) bb[j] = (_Float16)q[j < 8 ? j : j + 8];
        } else {
            #pragma unroll
            for (int j = 0; j < 16; ++j) bb[j] = (_Float16)0.0f;
        }
        acc = __builtin_amdgcn_wmma_f32_16x16x32_f16(
            /*neg_a=*/false, a, /*neg_b=*/false, bb,
            /*c_mod=*/(short)0, acc, /*reuse_a=*/false, /*reuse_b=*/false);
    }

    // D layout: VGPR i -> M = i + 8*half, N = lane&15
    int col = ntile * 16 + (lane & 15);
    if (col < N_CLASS) {
        float bias = b[col];
        #pragma unroll
        for (int i = 0; i < 8; ++i) {
            long long r = (long long)mtile * 16 + half * 8 + i;
            if (r < N) out[r * N_CLASS + col] = acc[i] + bias;
        }
    }
}

// ---------------- log_softmax over 40 classes, one wave32 per row -------------
__global__ void sgc_logsoftmax_kernel(float* __restrict__ out, int N) {
    int gid  = blockIdx.x * blockDim.x + threadIdx.x;
    int wave = gid >> 5;
    int lane = gid & 31;
    if (wave >= N) return;
    float* rowp = out + (long long)wave * N_CLASS;
    float v0 = rowp[lane];                                       // lanes 0..31 valid
    bool  hi = lane < (N_CLASS - 32);                            // lanes 0..7
    float v1 = hi ? rowp[lane + 32] : -3.402823466e38f;
    float mx = fmaxf(v0, v1);
    #pragma unroll
    for (int off = 16; off > 0; off >>= 1)
        mx = fmaxf(mx, __shfl_xor(mx, off, 32));
    float s = __expf(v0 - mx) + (hi ? __expf(v1 - mx) : 0.0f);
    #pragma unroll
    for (int off = 16; off > 0; off >>= 1)
        s += __shfl_xor(s, off, 32);
    float lse = mx + __logf(s);
    rowp[lane] = v0 - lse;
    if (hi) rowp[lane + 32] = v1 - lse;
}

extern "C" void kernel_launch(void* const* d_in, const int* in_sizes, int n_in,
                              void* d_out, int out_size, void* d_ws, size_t ws_size,
                              hipStream_t stream) {
    const float*     x   = (const float*)d_in[0];       // [N, 256]
    const long long* ei  = (const long long*)d_in[1];   // [2, E] int64
    const float*     ew  = (const float*)d_in[2];       // [E]
    const float*     W   = (const float*)d_in[3];       // [40, 256]
    const float*     b   = (const float*)d_in[4];       // [40]
    float*           out = (float*)d_out;               // [N, 40]

    int N = in_sizes[0] / N_FEAT;
    int E = in_sizes[2];
    const long long* row = ei;
    const long long* col = ei + E;

    // workspace carve-up
    char*  ws     = (char*)d_ws;
    size_t xbytes = (size_t)N * N_FEAT * sizeof(float);
    float* X1     = (float*)(ws);
    float* X2     = (float*)(ws + xbytes);
    float* nrm    = (float*)(ws + 2 * xbytes);
    float* dinv   = (float*)(ws + 2 * xbytes + (size_t)E * 4);
    float* selfn  = (float*)(ws + 2 * xbytes + (size_t)E * 4 + (size_t)N * 4);

    // --- normalization coefficients ---
    hipMemsetAsync(dinv, 0, (size_t)N * 4, stream);
    sgc_deg_kernel<<<(E + 255) / 256, 256, 0, stream>>>(col, ew, dinv, E);
    sgc_dinv_kernel<<<(N + 255) / 256, 256, 0, stream>>>(dinv, selfn, N);
    sgc_norm_kernel<<<(E + 255) / 256, 256, 0, stream>>>(row, col, ew, dinv, nrm, E);

    long long peThreads = (long long)E * 64;
    int peBlocks = (int)((peThreads + 255) / 256);
    long long psThreads = (long long)N * 64;
    int psBlocks = (int)((psThreads + 255) / 256);

    // --- hop 1: X1 = A_hat @ x ---
    hipMemsetAsync(X1, 0, xbytes, stream);
    sgc_prop_edge_kernel<<<peBlocks, 256, 0, stream>>>(row, col, nrm, x, X1, E);
    sgc_prop_self_kernel<<<psBlocks, 256, 0, stream>>>(selfn, x, X1, N);

    // --- hop 2: X2 = A_hat @ X1 ---
    hipMemsetAsync(X2, 0, xbytes, stream);
    sgc_prop_edge_kernel<<<peBlocks, 256, 0, stream>>>(row, col, nrm, X1, X2, E);
    sgc_prop_self_kernel<<<psBlocks, 256, 0, stream>>>(selfn, X1, X2, N);

    // --- classifier GEMM (WMMA) + bias, then log_softmax in place ---
    int mtiles = (N + 15) / 16;
    sgc_gemm_wmma_kernel<<<mtiles * 3, 32, 0, stream>>>(X2, W, b, out, N);
    sgc_logsoftmax_kernel<<<(int)(((long long)N * 32 + 255) / 256), 256, 0, stream>>>(out, N);
}
